// MaskedMoE_30897994727625
// MI455X (gfx1250) — compile-verified, matrix operand order checked by reading, not run on hardware
//
#include <hip/hip_runtime.h>
#include <stdint.h>

#define T_TOK 2048
#define D_EMB 1024
#define D_FF  4096
#define N_EXP 8

#define LDS_STRIDE 36   // shorts per row: 72B = 18 dwords, coprime-enough with 64 banks

typedef __attribute__((ext_vector_type(16))) __bf16 v16bf;
typedef __attribute__((ext_vector_type(8)))  float  v8f;

union FragBF { v16bf v; uint32_t u[8]; };

// Native bf16 convert (single v_cvt on gfx1250)
__device__ __forceinline__ unsigned short f2bf(float f) {
    return __builtin_bit_cast(unsigned short, static_cast<__bf16>(f));
}

__device__ __forceinline__ uint32_t pack2bf(float a, float b) {
    return (uint32_t)f2bf(a) | ((uint32_t)f2bf(b) << 16);
}

// gelu(x) = x * sigmoid(2*sqrt(2/pi)*(x + 0.044715x^3))  (tanh form, branch-free)
__device__ __forceinline__ float gelu_tanh(float x) {
    const float c = 0.7978845608028654f;          // sqrt(2/pi)
    float t2 = 2.0f * c * (x + 0.044715f * x * x * x);
    float em = __expf(-t2);                       // v_exp_f32
    return x * __builtin_amdgcn_rcpf(1.0f + em);  // v_rcp_f32
}

// k0 of the packed bf16 pair held in A/B fragment VGPR v, lane-half g
__device__ __forceinline__ int kmap(int v, int g) {
    return ((v < 4) ? (2 * v) : (2 * v + 8)) + 8 * g;
}

// ---------------- Router: logits, softmax, top-2, compaction ----------------
__global__ __launch_bounds__(256) void router_kernel(
    const float* __restrict__ x, const float* __restrict__ mask,
    const float* __restrict__ Wr, int* __restrict__ counts,
    int* __restrict__ tokList, float* __restrict__ wgtList)
{
    const int wave = threadIdx.x >> 5;
    const int lane = threadIdx.x & 31;
    const int t = blockIdx.x * 8 + wave;

    float acc[N_EXP];
#pragma unroll
    for (int e = 0; e < N_EXP; ++e) acc[e] = 0.f;

    const float* xr = x + (size_t)t * D_EMB;
    for (int k = lane; k < D_EMB; k += 32) {
        float xv = xr[k];
        const float* wr = Wr + (size_t)k * N_EXP;
#pragma unroll
        for (int e = 0; e < N_EXP; ++e) acc[e] += xv * wr[e];
    }
#pragma unroll
    for (int e = 0; e < N_EXP; ++e) {
#pragma unroll
        for (int off = 16; off >= 1; off >>= 1)
            acc[e] += __shfl_xor(acc[e], off);
    }

    if (lane == 0) {
        float p[N_EXP];
        float mx = -1e30f;
#pragma unroll
        for (int e = 0; e < N_EXP; ++e) {
            p[e] = acc[e] * mask[t * N_EXP + e];
            mx = fmaxf(mx, p[e]);
        }
        float s = 0.f;
#pragma unroll
        for (int e = 0; e < N_EXP; ++e) { p[e] = __expf(p[e] - mx); s += p[e]; }
        float inv = 1.f / s;
#pragma unroll
        for (int e = 0; e < N_EXP; ++e) p[e] *= inv;

        int i1 = 0;
#pragma unroll
        for (int e = 1; e < N_EXP; ++e) if (p[e] > p[i1]) i1 = e;
        int i2 = (i1 == 0) ? 1 : 0;
#pragma unroll
        for (int e = 0; e < N_EXP; ++e) if (e != i1 && p[e] > p[i2]) i2 = e;

        int s1 = atomicAdd(&counts[i1], 1);
        tokList[i1 * T_TOK + s1] = t;
        wgtList[i1 * T_TOK + s1] = p[i1];
        int s2 = atomicAdd(&counts[i2], 1);
        tokList[i2 * T_TOK + s2] = t;
        wgtList[i2 * T_TOK + s2] = p[i2];
    }
}

// ---------------- Prefix sums of per-expert counts ----------------
__global__ void offsets_kernel(const int* __restrict__ counts, int* __restrict__ rowStart)
{
    if (threadIdx.x == 0) {
        int s = 0;
        for (int e = 0; e < N_EXP; ++e) { rowStart[e] = s; s += counts[e]; }
        rowStart[N_EXP] = s;
    }
}

// ---------------- Grouped GEMM 1: h = gelu(x[tok] @ W1[e]) (bf16 out) ------
// Block tile 128(M) x 128(N), K-step 32. 8 waves, wave tile 32x64 (8 WMMA).
__global__ __launch_bounds__(256) void gemm1_kernel(
    const float* __restrict__ x, const float* __restrict__ W1,
    const int* __restrict__ counts, const int* __restrict__ rowStart,
    const int* __restrict__ tokList, unsigned short* __restrict__ h)
{
    const int mt = blockIdx.x;                 // M tile (128 rows)
    const int nt = blockIdx.y;                 // N tile (128 cols)
    const int e  = blockIdx.z;
    const int cnt = counts[e];
    if (mt * 128 >= cnt) return;               // routed-token sparsity
    const int rs = rowStart[e];
    const int nb = nt * 128;

    __shared__ unsigned short ldsA[128 * LDS_STRIDE];  // [row][k], bf16
    __shared__ unsigned short ldsB[128 * LDS_STRIDE];  // [n][k], bf16 (transposed)
    __shared__ int tokLds[128];

    const int tid = threadIdx.x;
    if (tid < 128) {
        int i = mt * 128 + tid;
        tokLds[tid] = tokList[e * T_TOK + ((i < cnt) ? i : 0)];
    }
    __syncthreads();

    const int wave  = tid >> 5;
    const int lane  = tid & 31;
    const int g     = lane >> 4;
    const int mloc  = lane & 15;
    const int waveM = wave & 3;                // 4 M groups of 32 rows
    const int waveN = wave >> 2;               // 2 N groups of 64 cols

    v8f z = {};
    v8f acc[8];                                // [am*4 + s]
#pragma unroll
    for (int i = 0; i < 8; ++i) acc[i] = z;

    const float* W1e = W1 + (size_t)e * D_EMB * D_FF;

    for (int kb = 0; kb < D_EMB; kb += 32) {
        __syncthreads();
        // A tile: 128x32 fp32 -> bf16, float4 loads, b64 LDS stores
#pragma unroll
        for (int it = 0; it < 4; ++it) {
            int chunk = tid + it * 256;                  // 1024 chunks of 4
            int r = chunk >> 3, c = (chunk & 7) * 4;
            float4 v = *(const float4*)&x[(size_t)tokLds[r] * D_EMB + kb + c];
            uint32_t p0 = pack2bf(v.x, v.y), p1 = pack2bf(v.z, v.w);
            uint64_t q = (uint64_t)p0 | ((uint64_t)p1 << 32);
            *(uint64_t*)&ldsA[r * LDS_STRIDE + c] = q;
        }
        // B tile: 32x128 fp32 -> bf16 transposed, float4 loads, 4x b16 stores
#pragma unroll
        for (int it = 0; it < 4; ++it) {
            int chunk = tid + it * 256;                  // 1024 chunks of 4 (along n)
            int k = chunk >> 5, nq = (chunk & 31) * 4;
            float4 v = *(const float4*)&W1e[(size_t)(kb + k) * D_FF + nb + nq];
            ldsB[(nq + 0) * LDS_STRIDE + k] = f2bf(v.x);
            ldsB[(nq + 1) * LDS_STRIDE + k] = f2bf(v.y);
            ldsB[(nq + 2) * LDS_STRIDE + k] = f2bf(v.z);
            ldsB[(nq + 3) * LDS_STRIDE + k] = f2bf(v.w);
        }
        __syncthreads();

        FragBF fa[2];
#pragma unroll
        for (int am = 0; am < 2; ++am) {
            const int arow = waveM * 32 + am * 16 + mloc;
#pragma unroll
            for (int v = 0; v < 8; ++v)
                fa[am].u[v] = *(const uint32_t*)&ldsA[arow * LDS_STRIDE + kmap(v, g)];
        }
#pragma unroll
        for (int s = 0; s < 4; ++s) {
            FragBF fb;
            const int bn = waveN * 64 + s * 16 + mloc;
#pragma unroll
            for (int v = 0; v < 8; ++v)
                fb.u[v] = *(const uint32_t*)&ldsB[bn * LDS_STRIDE + kmap(v, g)];
            acc[s]     = __builtin_amdgcn_wmma_f32_16x16x32_bf16(
                false, fa[0].v, false, fb.v, (short)0, acc[s], false, false);
            acc[4 + s] = __builtin_amdgcn_wmma_f32_16x16x32_bf16(
                false, fa[1].v, false, fb.v, (short)0, acc[4 + s], false, false);
        }
    }

#pragma unroll
    for (int am = 0; am < 2; ++am) {
#pragma unroll
        for (int s = 0; s < 4; ++s) {
#pragma unroll
            for (int r = 0; r < 8; ++r) {
                int row = waveM * 32 + am * 16 + r + 8 * g;
                int i = mt * 128 + row;
                if (i < cnt) {
                    int col = nb + waveN * 64 + s * 16 + mloc;
                    h[(size_t)(rs + i) * D_FF + col] =
                        f2bf(gelu_tanh(acc[am * 4 + s][r]));
                }
            }
        }
    }
}

// ---------------- Grouped GEMM 2: out[tok] += w * (h @ W2[e]) --------------
__global__ __launch_bounds__(256) void gemm2_kernel(
    const unsigned short* __restrict__ h, const float* __restrict__ W2,
    const int* __restrict__ counts, const int* __restrict__ rowStart,
    const int* __restrict__ tokList, const float* __restrict__ wgtList,
    float* __restrict__ out)
{
    const int mt = blockIdx.x;
    const int nt = blockIdx.y;                 // 0..7, 128 cols each
    const int e  = blockIdx.z;
    const int cnt = counts[e];
    if (mt * 128 >= cnt) return;
    const int rs = rowStart[e];
    const int nb = nt * 128;

    __shared__ unsigned short ldsA[128 * LDS_STRIDE];
    __shared__ unsigned short ldsB[128 * LDS_STRIDE];
    __shared__ int   tokLds[128];
    __shared__ float wLds[128];

    const int tid = threadIdx.x;
    if (tid < 128) {
        int i = mt * 128 + tid;
        int ri = (i < cnt) ? i : 0;
        tokLds[tid] = tokList[e * T_TOK + ri];
        wLds[tid]   = (i < cnt) ? wgtList[e * T_TOK + ri] : 0.f;
    }
    __syncthreads();

    const int wave  = tid >> 5;
    const int lane  = tid & 31;
    const int g     = lane >> 4;
    const int mloc  = lane & 15;
    const int waveM = wave & 3;
    const int waveN = wave >> 2;

    v8f z = {};
    v8f acc[8];
#pragma unroll
    for (int i = 0; i < 8; ++i) acc[i] = z;

    const float* W2e = W2 + (size_t)e * D_FF * D_EMB;

    for (int kb = 0; kb < D_FF; kb += 32) {
        __syncthreads();
        // A tile: straight bf16 copy, 8B global loads -> b64 LDS stores
#pragma unroll
        for (int it = 0; it < 4; ++it) {
            int chunk = tid + it * 256;
            int r = chunk >> 3, c = (chunk & 7) * 4;
            int i = mt * 128 + r;
            int ri = (i < cnt) ? i : 0;
            uint64_t q = *(const uint64_t*)&h[(size_t)(rs + ri) * D_FF + kb + c];
            *(uint64_t*)&ldsA[r * LDS_STRIDE + c] = q;
        }
        // B tile: 32x128 fp32 -> bf16 transposed
#pragma unroll
        for (int it = 0; it < 4; ++it) {
            int chunk = tid + it * 256;
            int k = chunk >> 5, nq = (chunk & 31) * 4;
            float4 v = *(const float4*)&W2e[(size_t)(kb + k) * D_EMB + nb + nq];
            ldsB[(nq + 0) * LDS_STRIDE + k] = f2bf(v.x);
            ldsB[(nq + 1) * LDS_STRIDE + k] = f2bf(v.y);
            ldsB[(nq + 2) * LDS_STRIDE + k] = f2bf(v.z);
            ldsB[(nq + 3) * LDS_STRIDE + k] = f2bf(v.w);
        }
        __syncthreads();

        FragBF fa[2];
#pragma unroll
        for (int am = 0; am < 2; ++am) {
            const int arow = waveM * 32 + am * 16 + mloc;
#pragma unroll
            for (int v = 0; v < 8; ++v)
                fa[am].u[v] = *(const uint32_t*)&ldsA[arow * LDS_STRIDE + kmap(v, g)];
        }
#pragma unroll
        for (int s = 0; s < 4; ++s) {
            FragBF fb;
            const int bn = waveN * 64 + s * 16 + mloc;
#pragma unroll
            for (int v = 0; v < 8; ++v)
                fb.u[v] = *(const uint32_t*)&ldsB[bn * LDS_STRIDE + kmap(v, g)];
            acc[s]     = __builtin_amdgcn_wmma_f32_16x16x32_bf16(
                false, fa[0].v, false, fb.v, (short)0, acc[s], false, false);
            acc[4 + s] = __builtin_amdgcn_wmma_f32_16x16x32_bf16(
                false, fa[1].v, false, fb.v, (short)0, acc[4 + s], false, false);
        }
    }

#pragma unroll
    for (int am = 0; am < 2; ++am) {
#pragma unroll
        for (int s = 0; s < 4; ++s) {
#pragma unroll
            for (int r = 0; r < 8; ++r) {
                int row = waveM * 32 + am * 16 + r + 8 * g;
                int i = mt * 128 + row;
                if (i < cnt) {
                    int col = nb + waveN * 64 + s * 16 + mloc;
                    float v = acc[am * 4 + s][r] * wLds[row];
                    atomicAdd(&out[(size_t)tokLds[row] * D_EMB + col], v);
                }
            }
        }
    }
}

// ---------------- Host launch ----------------
extern "C" void kernel_launch(void* const* d_in, const int* in_sizes, int n_in,
                              void* d_out, int out_size, void* d_ws, size_t ws_size,
                              hipStream_t stream)
{
    const float* x    = (const float*)d_in[0];
    const float* mask = (const float*)d_in[1];
    const float* Wr   = (const float*)d_in[2];
    const float* W1   = (const float*)d_in[3];
    const float* W2   = (const float*)d_in[4];
    float* out = (float*)d_out;

    char* ws = (char*)d_ws;
    int*   counts   = (int*)(ws + 0);                       // 8 ints
    int*   rowStart = (int*)(ws + 64);                      // 9 ints
    int*   tokList  = (int*)(ws + 128);                     // 8*2048 ints
    float* wgtList  = (float*)(ws + 128 + N_EXP * T_TOK * 4);
    unsigned short* h = (unsigned short*)(ws + 256 * 1024); // 4096 x 4096 bf16

    hipMemsetAsync(counts, 0, N_EXP * sizeof(int), stream);
    hipMemsetAsync(out, 0, (size_t)out_size * sizeof(float), stream);

    router_kernel<<<T_TOK / 8, 256, 0, stream>>>(x, mask, Wr, counts, tokList, wgtList);
    offsets_kernel<<<1, 32, 0, stream>>>(counts, rowStart);
    gemm1_kernel<<<dim3(16, 32, N_EXP), 256, 0, stream>>>(x, W1, counts, rowStart, tokList, h);
    gemm2_kernel<<<dim3(16, 8, N_EXP), 256, 0, stream>>>(h, W2, counts, rowStart, tokList, wgtList, out);
}
